// EncoderLayer_30279519437335
// MI455X (gfx1250) — matrix operand order repaired and checked by reading, not compiled, not run on hardware
//
#include <hip/hip_runtime.h>
#include <hip/hip_bf16.h>
#include <math.h>

#define Bsz 32
#define Tsz 120
#define Gsz 25
#define Dsz 256

typedef __attribute__((ext_vector_type(16))) __bf16 v16bf;
typedef __attribute__((ext_vector_type(8)))  __bf16 v8bf;
typedef __attribute__((ext_vector_type(8)))  float  v8f;

// ---- LDS pool layout (elements) ----
// xs   : 6400 f32        x[b,t,:, :] tile (residual source)
// adjs : 640  f32        adjacency 25x25 (padded)
// hb   : 32*264 bf16     h -> attb -> x1b   (A-operand for WMMA, stride 264)
// qs   : 32*260 f32      Q -> z -> z2       (stride 260)
// ks   : 32*260 f32      K
// vs   : 32*260 f32      V -> x1 (fp32)
// scw  : 1600 f32        scores / softmax weights [g][h1][h2]
#define HB_STRIDE 264
#define QV_STRIDE 260
#define SMEM_BYTES ((6400 + 640) * 4 + 32 * HB_STRIDE * 2 + (3 * 32 * QV_STRIDE + 1600) * 4)

__global__ void convert_weights_bf16(const float* __restrict__ Wq,
                                     const float* __restrict__ Wk,
                                     const float* __restrict__ Wv,
                                     const float* __restrict__ Wo,
                                     const float* __restrict__ Wfc,
                                     __bf16* __restrict__ wbf) {
    int i = blockIdx.x * 256 + threadIdx.x;
    if (i < 65536) {
        wbf[0 * 65536 + i] = (__bf16)Wq[i];
        wbf[1 * 65536 + i] = (__bf16)Wk[i];
        wbf[2 * 65536 + i] = (__bf16)Wv[i];
        wbf[3 * 65536 + i] = (__bf16)Wo[i];
        wbf[4 * 65536 + i] = (__bf16)Wfc[i];
    }
}

__device__ __forceinline__ float wave_sum(float v) {
    #pragma unroll
    for (int off = 16; off > 0; off >>= 1) v += __shfl_xor(v, off, 32);
    return v;
}

// D(16x16,f32) = A(16x32,bf16 from LDS rows, stride HB_STRIDE) x B(32x16,bf16 = W^T) , K=256
__device__ __forceinline__ v8f gemm_tile(const __bf16* __restrict__ hb,
                                         const __bf16* __restrict__ W,
                                         int mt, int nt, int lane) {
    const int mr = lane & 15;
    const int hi = lane >> 4;
    const __bf16* arow = hb + (mt * 16 + mr) * HB_STRIDE;
    const __bf16* brow = W + (nt * 16 + mr) * 256 + hi * 16;
    v8f acc = {};
    #pragma unroll
    for (int kt = 0; kt < 8; ++kt) {
        const int kb = kt * 32;
        // A frag: lanes 0-15 -> K = kb+{0..15}; lanes 16-31 -> K = kb+{16..31}
        v8bf alo = *(const v8bf*)(arow + kb + hi * 16);
        v8bf ahi = *(const v8bf*)(arow + kb + hi * 16 + 8);
        v16bf a;
        #pragma unroll
        for (int i = 0; i < 8; ++i) { a[i] = alo[i]; a[i + 8] = ahi[i]; }
        // B frag: element e -> K = kb + hi*16 + e, N = nt*16 + (lane&15); contiguous 32B
        v16bf b = *(const v16bf*)(brow + kb);
        acc = __builtin_amdgcn_wmma_f32_16x16x32_bf16(false, a, false, b,
                                                      (short)0, acc, false, false);
    }
    return acc;
}

__global__ void __launch_bounds__(256)
encoder_layer_fused(const float* __restrict__ x,
                    const float* __restrict__ adj,
                    const float* __restrict__ bq, const float* __restrict__ bk,
                    const float* __restrict__ bv, const float* __restrict__ bo,
                    const float* __restrict__ bfc,
                    const float* __restrict__ alpha, const float* __restrict__ beta,
                    const __bf16* __restrict__ wbf,
                    float* __restrict__ out) {
    extern __shared__ __align__(16) char smem[];
    float*  xs   = (float*)smem;                    // 6400
    float*  adjs = xs + 6400;                       // 640
    __bf16* hb   = (__bf16*)(adjs + 640);           // 32*264
    float*  qs   = (float*)(hb + 32 * HB_STRIDE);   // 32*260
    float*  ks   = qs + 32 * QV_STRIDE;
    float*  vs   = ks + 32 * QV_STRIDE;
    float*  scw  = vs + 32 * QV_STRIDE;             // 1600

    const int tid  = threadIdx.x;
    const int wid  = tid >> 5;
    const int lane = tid & 31;
    const int bt   = blockIdx.x;                    // (b*T + t)

    const __bf16* wq  = wbf;
    const __bf16* wk  = wbf + 65536;
    const __bf16* wv  = wbf + 2 * 65536;
    const __bf16* wo  = wbf + 3 * 65536;
    const __bf16* wfc = wbf + 4 * 65536;

    // pull weight lines toward caches (global_prefetch_b8)
    #pragma unroll
    for (int r = 0; r < 5; ++r) __builtin_prefetch(wbf + r * 65536 + tid * 256, 0, 0);

    // ---- Stage 0: load x slice + adjacency; clear pad rows of hb ----
    const float* xg = x + (size_t)bt * (Gsz * Dsz);
    for (int i = tid; i < Gsz * Dsz; i += 256) xs[i] = xg[i];
    for (int i = tid; i < Gsz * Gsz; i += 256) adjs[i] = adj[i];
    for (int i = tid; i < 7 * HB_STRIDE; i += 256) hb[Gsz * HB_STRIDE + i] = (__bf16)0.0f;
    __syncthreads();

    // ---- Stage 1: h = A @ x (per-(b,t) [25x25]x[25x256]) -> bf16 ----
    {
        const int d = tid;                          // 0..255
        for (int g = 0; g < Gsz; ++g) {
            float acc = 0.f;
            #pragma unroll 5
            for (int n = 0; n < Gsz; ++n) acc += adjs[g * Gsz + n] * xs[n * Dsz + d];
            hb[g * HB_STRIDE + d] = (__bf16)acc;
        }
    }
    __syncthreads();

    // ---- Stage 2: Q,K,V = h @ W^T + b  (96 WMMA tile-jobs over 8 waves) ----
    for (int i = 0; i < 12; ++i) {
        const int j   = wid * 12 + i;               // 0..95
        const int p   = j >> 5;                     // 0:Q 1:K 2:V
        const int rem = j & 31;
        const int mt  = rem >> 4, nt = rem & 15;
        const __bf16* W  = (p == 0) ? wq : (p == 1) ? wk : wv;
        const float*  bp = (p == 0) ? bq : (p == 1) ? bk : bv;
        float*        dp = (p == 0) ? qs : (p == 1) ? ks : vs;
        v8f acc = gemm_tile(hb, W, mt, nt, lane);
        const int nn    = nt * 16 + (lane & 15);
        const int mbase = mt * 16 + (lane >> 4) * 8;
        const float bb  = bp[nn];
        #pragma unroll
        for (int r = 0; r < 8; ++r) dp[(mbase + r) * QV_STRIDE + nn] = acc[r] + bb;
    }
    __syncthreads();

    // ---- Stage 3a: scores[g,h1,h2] = Q.K^T / sqrt(HD) ----
    const float scale = 0.17677669529663687f;       // 1/sqrt(32)
    for (int idx = tid; idx < 1600; idx += 256) {
        const int g = idx >> 6, rem = idx & 63, h1 = rem >> 3, h2 = rem & 7;
        const float* qrow = qs + g * QV_STRIDE + h1 * 32;
        const float* krow = ks + g * QV_STRIDE + h2 * 32;
        float s = 0.f;
        #pragma unroll
        for (int d = 0; d < 32; ++d) s += qrow[d] * krow[d];
        scw[idx] = s * scale;
    }
    __syncthreads();

    // ---- Stage 3b: softmax over G (torch dim=1 quirk) ----
    if (tid < 64) {
        float m = -3.0e38f;
        for (int g = 0; g < Gsz; ++g) m = fmaxf(m, scw[g * 64 + tid]);
        float ssum = 0.f;
        for (int g = 0; g < Gsz; ++g) {
            float e = __expf(scw[g * 64 + tid] - m);
            scw[g * 64 + tid] = e;
            ssum += e;
        }
        const float inv = 1.0f / ssum;
        for (int g = 0; g < Gsz; ++g) scw[g * 64 + tid] *= inv;
    }
    __syncthreads();

    // ---- Stage 3c: att = w @ V -> bf16 into hb (as [g][h1*32+hd]) ----
    for (int idx = tid; idx < Gsz * Dsz; idx += 256) {
        const int g = idx >> 8, d = idx & 255, h1 = d >> 5, hd = d & 31;
        const float* wrow = scw + g * 64 + h1 * 8;
        const float* vcol = vs + g * QV_STRIDE + hd;
        float a = 0.f;
        #pragma unroll
        for (int h2 = 0; h2 < 8; ++h2) a += wrow[h2] * vcol[h2 * 32];
        hb[g * HB_STRIDE + d] = (__bf16)a;
    }
    __syncthreads();

    // ---- Stage 4: z = x + att @ Wo^T + bo  -> qs ----
    for (int i = 0; i < 4; ++i) {
        const int j = wid * 4 + i;
        const int mt = j >> 4, nt = j & 15;
        v8f acc = gemm_tile(hb, wo, mt, nt, lane);
        const int nn    = nt * 16 + (lane & 15);
        const int mbase = mt * 16 + (lane >> 4) * 8;
        const float bb  = bo[nn];
        #pragma unroll
        for (int r = 0; r < 8; ++r) {
            const int m = mbase + r;
            const float resid = (m < Gsz) ? xs[m * Dsz + nn] : 0.f;
            qs[m * QV_STRIDE + nn] = acc[r] + bb + resid;
        }
    }
    __syncthreads();

    // ---- Stage 5: x1 = LN(z) -> vs (fp32) and hb (bf16) ----
    for (int g = wid; g < Gsz; g += 8) {
        const float* zr = qs + g * QV_STRIDE;
        float s = 0.f, sq = 0.f, zv[8];
        #pragma unroll
        for (int c = 0; c < 8; ++c) {
            float z = zr[lane + c * 32];
            zv[c] = z; s += z; sq += z * z;
        }
        s = wave_sum(s); sq = wave_sum(sq);
        const float mu  = s * (1.0f / 256.0f);
        float var = (sq - 256.0f * mu * mu) * (1.0f / 255.0f);   // ddof=1
        var = fmaxf(var, 0.f);
        const float inv = 1.0f / (sqrtf(var) + 1e-6f);
        #pragma unroll
        for (int c = 0; c < 8; ++c) {
            const int d = lane + c * 32;
            const float xv = alpha[d] * (zv[c] - mu) * inv + beta[d];
            vs[g * QV_STRIDE + d] = xv;
            hb[g * HB_STRIDE + d] = (__bf16)xv;
        }
    }
    __syncthreads();

    // ---- Stage 6: z2 = x1 + relu(x1 @ Wfc^T + bfc) -> qs ----
    for (int i = 0; i < 4; ++i) {
        const int j = wid * 4 + i;
        const int mt = j >> 4, nt = j & 15;
        v8f acc = gemm_tile(hb, wfc, mt, nt, lane);
        const int nn    = nt * 16 + (lane & 15);
        const int mbase = mt * 16 + (lane >> 4) * 8;
        const float bb  = bfc[nn];
        #pragma unroll
        for (int r = 0; r < 8; ++r) {
            const int m = mbase + r;
            const float fc  = fmaxf(acc[r] + bb, 0.f);
            const float x1v = (m < Gsz) ? vs[m * QV_STRIDE + nn] : 0.f;
            qs[m * QV_STRIDE + nn] = x1v + fc;
        }
    }
    __syncthreads();

    // ---- Stage 7: out = LN(z2) -> global ----
    float* outg = out + (size_t)bt * (Gsz * Dsz);
    for (int g = wid; g < Gsz; g += 8) {
        const float* zr = qs + g * QV_STRIDE;
        float s = 0.f, sq = 0.f, zv[8];
        #pragma unroll
        for (int c = 0; c < 8; ++c) {
            float z = zr[lane + c * 32];
            zv[c] = z; s += z; sq += z * z;
        }
        s = wave_sum(s); sq = wave_sum(sq);
        const float mu  = s * (1.0f / 256.0f);
        float var = (sq - 256.0f * mu * mu) * (1.0f / 255.0f);
        var = fmaxf(var, 0.f);
        const float inv = 1.0f / (sqrtf(var) + 1e-6f);
        #pragma unroll
        for (int c = 0; c < 8; ++c) {
            const int d = lane + c * 32;
            outg[g * Dsz + d] = alpha[d] * (zv[c] - mu) * inv + beta[d];
        }
    }
}

extern "C" void kernel_launch(void* const* d_in, const int* in_sizes, int n_in,
                              void* d_out, int out_size, void* d_ws, size_t ws_size,
                              hipStream_t stream) {
    (void)in_sizes; (void)n_in; (void)out_size; (void)ws_size;
    const float* x     = (const float*)d_in[0];
    const float* adj   = (const float*)d_in[1];
    const float* Wq    = (const float*)d_in[2];
    const float* bq    = (const float*)d_in[3];
    const float* Wk    = (const float*)d_in[4];
    const float* bk    = (const float*)d_in[5];
    const float* Wv    = (const float*)d_in[6];
    const float* bv    = (const float*)d_in[7];
    const float* Wo    = (const float*)d_in[8];
    const float* bo    = (const float*)d_in[9];
    const float* Wfc   = (const float*)d_in[10];
    const float* bfc   = (const float*)d_in[11];
    const float* alpha = (const float*)d_in[12];
    const float* beta  = (const float*)d_in[13];

    __bf16* wbf = (__bf16*)d_ws;   // 5 * 65536 bf16 = 640 KB

    convert_weights_bf16<<<256, 256, 0, stream>>>(Wq, Wk, Wv, Wo, Wfc, wbf);

    encoder_layer_fused<<<Bsz * Tsz, 256, SMEM_BYTES, stream>>>(
        x, adj, bq, bk, bv, bo, bfc, alpha, beta, wbf, (float*)d_out);
}
